// LSTM_43301860278766
// MI455X (gfx1250) — matrix operand order repaired and checked by reading, not compiled, not run on hardware
//
#include <hip/hip_runtime.h>
#include <hip/hip_bf16.h>

typedef __attribute__((ext_vector_type(16))) __bf16 v16bf;
typedef __attribute__((ext_vector_type(8)))  float  v8f;

#define T_STEPS 2048
#define BATCH   64
#define DIN     256
#define DLAT    256
#define DOUT    256
#define KDIM    512      // DIN + DLAT
#define NWG     16       // workgroups; each owns 16 latent dims
#define NTHR    256      // 8 wave32s

#define COMB_LD 520      // bf16 per comb row (512 + 8 pad); 1040B rows (16B aligned)
#define W_LD    24       // bf16 per weight k-row (16 used + 8 pad); 48B rows (16B aligned)
#define G_LD    17       // f32 per row for gate scratch / C state

// ---- dynamic LDS carve-up (bytes) ----
#define OFF_W4    0
#define SZ_W4     (4*KDIM*W_LD*2)              // 98304  gate weights, B layout, bf16
#define OFF_WO    (OFF_W4 + SZ_W4)             // 98304
#define SZ_WO     (DLAT*W_LD*2)                // 12288  Wout slice, B layout
#define OFF_COMB  (OFF_WO + SZ_WO)             // 110592 (16B aligned)
#define SZ_COMB   (BATCH*COMB_LD*2)            // 66560  comb = [x_t | h_{t-1}] bf16
#define OFF_GATES (OFF_COMB + SZ_COMB)         // 177152
#define SZ_GATES  (4*BATCH*G_LD*4)             // 17408  gate pre-activations f32
#define OFF_CST   (OFF_GATES + SZ_GATES)       // 194560
#define SZ_CST    (BATCH*G_LD*4)               // 4352   cell state slice f32
#define OFF_BIAS  (OFF_CST + SZ_CST)           // 198912
#define SZ_BIAS   (4*16*4)
#define OFF_BOUT  (OFF_BIAS + SZ_BIAS)
#define SZ_BOUT   (16*4)
#define SMEM_TOTAL (OFF_BOUT + SZ_BOUT)        // 199232 < 320KB WGP LDS

__device__ inline unsigned short f2bf(float f) {          // RNE fp32 -> bf16
  unsigned u = __float_as_uint(f);
  return (unsigned short)((u + 0x7FFFu + ((u >> 16) & 1u)) >> 16);
}
__device__ inline unsigned pack2(float a, float b) {
  return (unsigned)f2bf(a) | ((unsigned)f2bf(b) << 16);
}
__device__ inline float sigf(float x) { return 1.0f / (1.0f + __expf(-x)); }
__device__ inline float tanhfast(float x) {
  x = fminf(15.0f, fmaxf(-15.0f, x));
  float e = __expf(2.0f * x);
  return (e - 1.0f) / (e + 1.0f);
}

__device__ inline v8f wmma_bf16(v16bf a, v16bf b, v8f c) {
  return __builtin_amdgcn_wmma_f32_16x16x32_bf16(false, a, false, b, (short)0, c,
                                                 false, false);
}

// A fragment: 16x32 bf16 tile (documented 16-bit A layout).
//   lane L: M = L&15; half = L>>4 adds +8 to K.
//   dwords 0..3 = K pairs {0..7}+8*half ; dwords 4..7 = K pairs {16..23}+8*half.
// Both 4-dword runs are contiguous and 16B aligned -> two ds_load_b128.
__device__ inline v16bf load_a(const unsigned short* COMB, int mtile, int kbase, int lane) {
  const uint4* row =
      (const uint4*)(COMB + (size_t)(mtile * 16 + (lane & 15)) * COMB_LD);
  int q0 = (kbase >> 3) + (lane >> 4);       // uint4 index: kbase/8 + half
  union { v16bf v; uint4 q[2]; } r;
  r.q[0] = row[q0];
  r.q[1] = row[q0 + 2];
  return r.v;
}

// B fragment: 32x16 bf16 tile; lane L holds column N=L&15, K = {0..7,16..23}+8*half (strided gather).
__device__ inline v16bf load_b(const unsigned short* W, int ks, int lane) {
  const unsigned short* p = W + (size_t)(ks * 32 + ((lane >> 4) << 3)) * W_LD + (lane & 15);
  union { v16bf v; unsigned short u[16]; } r;
#pragma unroll
  for (int s = 0; s < 8; ++s) {
    r.u[s] = p[(size_t)s * W_LD];
    r.u[8 + s] = p[(size_t)(16 + s) * W_LD];
  }
  return r.v;
}

// Stage comb = [x_t (bf16) | h_{t-1} (bf16)] into LDS. t==T_STEPS: h-part only.
__device__ inline void stage_comb(int t, const float* __restrict__ x,
                                  const float* __restrict__ h0,
                                  const float* __restrict__ out_h,
                                  unsigned short* COMB, int tid) {
  int b = tid >> 2;
  int kb = (tid & 3) * 128;
  const float* src;
  if (kb < 256) {
    if (t >= T_STEPS) return;
    src = x + ((size_t)t * BATCH + b) * DIN + kb;
    if (t + 1 < T_STEPS) __builtin_prefetch(src + (size_t)BATCH * DIN, 0, 1);
  } else {
    int j = kb - 256;
    src = (t == 0) ? (h0 + (size_t)b * DLAT + j)
                   : (out_h + ((size_t)(t - 1) * BATCH + b) * DLAT + j);
  }
  uint2* dst = (uint2*)(COMB + (size_t)b * COMB_LD + kb);   // 8B aligned
  const float4* s4 = (const float4*)src;
#pragma unroll 8
  for (int i = 0; i < 32; ++i) {
    float4 v = s4[i];
    uint2 p;
    p.x = pack2(v.x, v.y);
    p.y = pack2(v.z, v.w);
    dst[i] = p;                                              // ds_store_b64
  }
}

__device__ inline void outproj(const unsigned short* COMB, const unsigned short* WO,
                               const float* BOUT, float* __restrict__ dst,
                               int j0, int wv, int lane) {
  if (wv >= 4) return;                       // wave-uniform: EXEC stays all-ones
  v8f acc = {};
#pragma unroll
  for (int ks = 0; ks < 8; ++ks) {
    v16bf a = load_a(COMB, wv, 256 + ks * 32, lane);
    v16bf bb = load_b(WO, ks, lane);
    acc = wmma_bf16(a, bb, acc);
  }
  int n = lane & 15;
  int b0 = wv * 16 + (lane >> 4) * 8;
  float bbias = BOUT[n];
#pragma unroll
  for (int r = 0; r < 8; ++r)
    dst[(size_t)(b0 + r) * DOUT + j0 + n] = acc[r] + bbias;
}

__global__ void __launch_bounds__(NTHR, 1)
lstm_persistent(const float* __restrict__ x, const float* __restrict__ h0,
                const float* __restrict__ C0,
                const float* __restrict__ Wf, const float* __restrict__ bf_,
                const float* __restrict__ Wi, const float* __restrict__ bi_,
                const float* __restrict__ Wo, const float* __restrict__ bo_,
                const float* __restrict__ Wc, const float* __restrict__ bc_,
                const float* __restrict__ Wout, const float* __restrict__ bout_,
                float* __restrict__ out_y, float* __restrict__ out_h,
                float* __restrict__ out_cf, unsigned* __restrict__ bar) {
  extern __shared__ char smem[];
  unsigned short* W4   = (unsigned short*)(smem + OFF_W4);
  unsigned short* WO   = (unsigned short*)(smem + OFF_WO);
  unsigned short* COMB = (unsigned short*)(smem + OFF_COMB);
  float* GATES = (float*)(smem + OFF_GATES);
  float* CST   = (float*)(smem + OFF_CST);
  float* BIAS  = (float*)(smem + OFF_BIAS);
  float* BOUT  = (float*)(smem + OFF_BOUT);

  const int tid  = threadIdx.x;
  const int wg   = blockIdx.x;
  const int lane = tid & 31;
  const int wv   = tid >> 5;
  const int j0   = wg * 16;

  // ---- one-time: stage weight slices into LDS (bf16, WMMA-B layout) ----
  {
    int row = tid >> 2;            // 0..63 = (gate, j)
    int g = row >> 4, j = row & 15;
    int kb = (tid & 3) * 128;
    const float* src;
    if (g == 0)      src = Wf;
    else if (g == 1) src = Wi;
    else if (g == 2) src = Wo;
    else             src = Wc;
    src += (size_t)(j0 + j) * KDIM + kb;
    unsigned short* dstb = W4 + (size_t)g * KDIM * W_LD + j;
#pragma unroll 4
    for (int k = 0; k < 128; ++k) dstb[(size_t)(kb + k) * W_LD] = f2bf(src[k]);
  }
  {
    int j = tid >> 4;              // 0..15
    int kb = (tid & 15) * 16;
    const float* src = Wout + (size_t)(j0 + j) * DLAT + kb;
#pragma unroll
    for (int k = 0; k < 16; ++k) WO[(size_t)(kb + k) * W_LD + j] = f2bf(src[k]);
  }
  if (tid < 64) {
    int g = tid >> 4, j = tid & 15;
    const float* bg = (g == 0) ? bf_ : (g == 1) ? bi_ : (g == 2) ? bo_ : bc_;
    BIAS[tid] = bg[j0 + j];
  }
  if (tid < 16) BOUT[tid] = bout_[j0 + tid];
#pragma unroll
  for (int i = 0; i < 4; ++i) {
    int e = tid + NTHR * i;
    int b = e >> 4, j = e & 15;
    CST[b * G_LD + j] = C0[(size_t)b * DLAT + j0 + j];
  }
  __syncthreads();

  // ---- sequential scan ----
  for (int t = 0; t < T_STEPS; ++t) {
    stage_comb(t, x, h0, out_h, COMB, tid);
    __syncthreads();

    if (t > 0)
      outproj(COMB, WO, BOUT, out_y + (size_t)(t - 1) * BATCH * DOUT, j0, wv, lane);

    // gate GEMMs: wave wv -> gate g = wv>>1, M-tiles {m0, m0+1}
    {
      int g = wv >> 1;
      int m0 = (wv & 1) * 2;
      const unsigned short* Wg = W4 + (size_t)g * KDIM * W_LD;
      v8f acc0 = {};
      v8f acc1 = {};
#pragma unroll
      for (int ks = 0; ks < 16; ++ks) {
        v16bf bb = load_b(Wg, ks, lane);
        v16bf a0 = load_a(COMB, m0, ks * 32, lane);
        v16bf a1 = load_a(COMB, m0 + 1, ks * 32, lane);
        acc0 = wmma_bf16(a0, bb, acc0);
        acc1 = wmma_bf16(a1, bb, acc1);
      }
      int n = lane & 15;
      int bh = (lane >> 4) * 8;
      float* gb = GATES + (size_t)g * BATCH * G_LD;
#pragma unroll
      for (int r = 0; r < 8; ++r) {
        gb[(m0 * 16 + bh + r) * G_LD + n]       = acc0[r];
        gb[((m0 + 1) * 16 + bh + r) * G_LD + n] = acc1[r];
      }
    }
    __syncthreads();

    // fused elementwise: gates -> C_new, h_new (this WG's 16 latent dims)
#pragma unroll
    for (int i = 0; i < 4; ++i) {
      int e = tid + NTHR * i;
      int b = e >> 4, j = e & 15;
      float pf = GATES[(0 * BATCH + b) * G_LD + j] + BIAS[j];
      float pi = GATES[(1 * BATCH + b) * G_LD + j] + BIAS[16 + j];
      float po = GATES[(2 * BATCH + b) * G_LD + j] + BIAS[32 + j];
      float pc = GATES[(3 * BATCH + b) * G_LD + j] + BIAS[48 + j];
      float f  = sigf(pf), ig = sigf(pi), o = sigf(po), ct = tanhfast(pc);
      float Cn = f * CST[b * G_LD + j] + ig * ct;
      float hn = o * tanhfast(Cn);
      CST[b * G_LD + j] = Cn;
      out_h[((size_t)t * BATCH + b) * DLAT + j0 + j] = hn;
    }

    // device-scope barrier: publish h_t, wait for all 16 WGs
    __threadfence();
    __syncthreads();
    if (tid == 0) {
      atomicAdd(bar, 1u);
      unsigned target = (unsigned)(NWG * (t + 1));
      while (*(volatile unsigned*)bar < target) __builtin_amdgcn_s_sleep(1);
    }
    __syncthreads();
    __threadfence();
  }

  // epilogue: out_{T-1} and final cell state
  stage_comb(T_STEPS, x, h0, out_h, COMB, tid);
  __syncthreads();
  outproj(COMB, WO, BOUT, out_y + (size_t)(T_STEPS - 1) * BATCH * DOUT, j0, wv, lane);
#pragma unroll
  for (int i = 0; i < 4; ++i) {
    int e = tid + NTHR * i;
    int b = e >> 4, j = e & 15;
    out_cf[(size_t)b * DLAT + j0 + j] = CST[b * G_LD + j];
  }
}

__global__ void ws_init_kernel(unsigned* p) {
  if (threadIdx.x < 64) p[threadIdx.x] = 0u;
}

extern "C" void kernel_launch(void* const* d_in, const int* in_sizes, int n_in,
                              void* d_out, int out_size, void* d_ws, size_t ws_size,
                              hipStream_t stream) {
  (void)in_sizes; (void)n_in; (void)out_size; (void)ws_size;
  const float* x    = (const float*)d_in[0];
  const float* h0   = (const float*)d_in[1];
  const float* C0   = (const float*)d_in[2];
  const float* Wf   = (const float*)d_in[3];
  const float* bf_  = (const float*)d_in[4];
  const float* Wi   = (const float*)d_in[5];
  const float* bi_  = (const float*)d_in[6];
  const float* Wo   = (const float*)d_in[7];
  const float* bo_  = (const float*)d_in[8];
  const float* Wc   = (const float*)d_in[9];
  const float* bc_  = (const float*)d_in[10];
  const float* Wout = (const float*)d_in[11];
  const float* bout = (const float*)d_in[12];

  float* out_y  = (float*)d_out;                                  // [T,B,256]
  float* out_h  = out_y + (size_t)T_STEPS * BATCH * DOUT;         // [T,B,256]
  float* out_cf = out_h + (size_t)T_STEPS * BATCH * DLAT;         // [B,256]
  unsigned* bar = (unsigned*)d_ws;

  ws_init_kernel<<<1, 64, 0, stream>>>(bar);

  hipFuncSetAttribute((const void*)lstm_persistent,
                      hipFuncAttributeMaxDynamicSharedMemorySize, SMEM_TOTAL);
  lstm_persistent<<<NWG, NTHR, SMEM_TOTAL, stream>>>(
      x, h0, C0, Wf, bf_, Wi, bi_, Wo, bo_, Wc, bc_, Wout, bout,
      out_y, out_h, out_cf, bar);
}
